// DFlashModel_64484638982500
// MI455X (gfx1250) — matrix operand order repaired and checked by reading, not compiled, not run on hardware
//
#include <hip/hip_runtime.h>
#include <hip/hip_bf16.h>
#include <math.h>

#define DEV __device__ __forceinline__

typedef __bf16 bf16;
typedef __attribute__((ext_vector_type(16))) __bf16 v16bf;
typedef __attribute__((ext_vector_type(8)))  float  v8f;

constexpr int S  = 2048, Nb = 128, BS = 16, D = 512, H = 8, V = 32000;
constexpr int Q  = Nb * BS;      // 2048 draft queries
constexpr int KV = S + Q;        // 4096 keys/values
constexpr int DH = D / H;        // 64 head dim
constexpr int VB = 256;          // vocab cols per logits block
constexpr int NVB = V / VB;      // 125 vocab blocks
constexpr int MASK_ID = 3;

template <bool B> struct BoolC { static constexpr bool value = B; };

// ---------------- scalar bf16 helpers (bit-exact RNE, no libcalls) ----------
DEV bf16 f2bf(float f) {
  unsigned u = __float_as_uint(f);
  unsigned short h = (unsigned short)((u + 0x7FFFu + ((u >> 16) & 1u)) >> 16);
  return *reinterpret_cast<bf16*>(&h);
}
DEV float bf2f(bf16 x) {
  unsigned short h = *reinterpret_cast<unsigned short*>(&x);
  return __uint_as_float(((unsigned)h) << 16);
}
DEV v8f zero8() { v8f z = {0.f,0.f,0.f,0.f,0.f,0.f,0.f,0.f}; return z; }
DEV v8f wmma_bf16(v16bf a, v16bf b, v8f c) {
  return __builtin_amdgcn_wmma_f32_16x16x32_bf16(false, a, false, b, (short)0, c, false, false);
}

// ---------------- WMMA fragment loaders (contiguous k order: lane half h holds k 16h..16h+15) ----------------
// A (16x32): lanes 0-15 row M=lane, K 0-15; lanes 16-31 K 16-31.
// Row-major A -> one contiguous 32B run per lane.
DEV v16bf load_a16x32(const bf16* A, int lda, int m0, int k0) {
  int lane = threadIdx.x & 31;
  const bf16* row = A + (size_t)(m0 + (lane & 15)) * lda + k0 + ((lane >> 4) << 4);
  v16bf f;
#pragma unroll
  for (int v = 0; v < 8; ++v) {
    int k = 2 * v;
    f[2 * v]     = row[k];
    f[2 * v + 1] = row[k + 1];
  }
  return f;
}
// B (32x16) from an N-major (transposed) matrix BT: B[kk][n] = BT[n*ldb + kk].
// One contiguous 32B run per lane -> global_load_b128 x2.
DEV v16bf load_bn(const bf16* BT, int ldb, int k0, int n0) {
  int lane = threadIdx.x & 31;
  const bf16* row = BT + (size_t)(n0 + (lane & 15)) * ldb + k0 + ((lane >> 4) << 4);
  v16bf f;
#pragma unroll
  for (int v = 0; v < 8; ++v) { f[2 * v] = row[2 * v]; f[2 * v + 1] = row[2 * v + 1]; }
  return f;
}
// Variant clamping the n (row) index: draft-tail K^T reads past KV.
DEV v16bf load_bn_clampN(const bf16* BT, int ldb, int k0, int n0, int nmax) {
  int lane = threadIdx.x & 31;
  int n = n0 + (lane & 15); n = n < nmax ? n : nmax - 1;
  const bf16* row = BT + (size_t)n * ldb + k0 + ((lane >> 4) << 4);
  v16bf f;
#pragma unroll
  for (int v = 0; v < 8; ++v) { f[2 * v] = row[2 * v]; f[2 * v + 1] = row[2 * v + 1]; }
  return f;
}
// Variant clamping the k (element) index: draft-tail V^T reads past KV.
DEV v16bf load_bn_clampK(const bf16* BT, int ldb, int k0, int n0, int kmax) {
  int lane = threadIdx.x & 31;
  const bf16* row = BT + (size_t)(n0 + (lane & 15)) * ldb;
  int kb = k0 + ((lane >> 4) << 4);
  v16bf f;
#pragma unroll
  for (int v = 0; v < 8; ++v) {
    int ka = kb + 2 * v; ka = ka < kmax ? ka : kmax - 1;
    int kc = kb + 2 * v + 1; kc = kc < kmax ? kc : kmax - 1;
    f[2 * v] = row[ka]; f[2 * v + 1] = row[kc];
  }
  return f;
}

// ---------------- prep kernels ---------------------------------------------
__global__ void cvt_f32_bf16(const float* __restrict__ in, bf16* __restrict__ out, int n) {
  int i = blockIdx.x * blockDim.x + threadIdx.x;
  if (i < n) out[i] = f2bf(in[i]);
}

// K x N f32  ->  N x K bf16, 32x32 tiles staged in LDS (both sides coalesced)
__global__ void cvt_transpose(const float* __restrict__ in, bf16* __restrict__ out,
                              int K, int N) {
  __shared__ float t[32][33];
  int kb = blockIdx.x << 5, nb = blockIdx.y << 5;
  int tx = threadIdx.x & 31, ty = threadIdx.x >> 5;   // 256 thr = 32x8
#pragma unroll
  for (int i = ty; i < 32; i += 8)
    t[i][tx] = in[(size_t)(kb + i) * N + nb + tx];
  __syncthreads();
#pragma unroll
  for (int i = ty; i < 32; i += 8)
    out[(size_t)(nb + i) * K + kb + tx] = f2bf(t[tx][i]);
}

__global__ void embed_noise(const int* __restrict__ ids, const int* __restrict__ anchors,
                            const unsigned char* __restrict__ keep,
                            const float* __restrict__ embed, bf16* __restrict__ kv_noise) {
  int i = blockIdx.x * blockDim.x + threadIdx.x;   // over Q*D
  if (i >= Q * D) return;
  int q = i / D, d = i - q * D;
  int off = q & (BS - 1), n = q >> 4;
  int tok = MASK_ID;
  if (off == 0) {
    int a = anchors[n]; a = a < 0 ? 0 : (a > S - 1 ? S - 1 : a);
    tok = keep[n] ? ids[a] : MASK_ID;
  }
  kv_noise[i] = f2bf(embed[(size_t)tok * D + d]);
}

__global__ void targets_weights(const int* __restrict__ ids, const float* __restrict__ lmask,
                                const int* __restrict__ anchors, const unsigned char* __restrict__ keep,
                                int* __restrict__ target, float* __restrict__ weight) {
  int r = blockIdx.x * blockDim.x + threadIdx.x;
  if (r >= Q) return;
  int n = r >> 4, off = r & 15;
  int label = anchors[n] + off;
  int valid = label < S;
  int safe  = label < 0 ? 0 : (label >= S ? S - 1 : label);
  target[r] = ids[safe];
  weight[r] = (keep[n] ? 1.f : 0.f) * (valid ? 1.f : 0.f) * (off > 0 ? 1.f : 0.f) * lmask[safe];
}

// ------ WMMA GEMM, 16x64 per wave: C(MxN) = A(MxK) @ BT(NxK)^T, optional C^T --
template <bool TC>
__global__ void gemm_bf16(const bf16* __restrict__ A, const bf16* __restrict__ BT,
                          bf16* __restrict__ C, int M, int Nn, int K) {
  int wave = (blockIdx.x * blockDim.x + threadIdx.x) >> 5;
  int tilesN = Nn >> 6;
  int tm = (wave / tilesN) << 4;
  int tn = (wave % tilesN) << 6;
  if (tm >= M) return;
  v8f a0 = zero8(), a1 = zero8(), a2 = zero8(), a3 = zero8();
  for (int k = 0; k < K; k += 32) {
    if (k + 32 < K) {
      __builtin_prefetch(A  + (size_t)(tm + (threadIdx.x & 15)) * K + k + 32, 0, 3);
      __builtin_prefetch(BT + (size_t)(tn + (threadIdx.x & 31)) * K + k + 32, 0, 3);
      __builtin_prefetch(BT + (size_t)(tn + 32 + (threadIdx.x & 31)) * K + k + 32, 0, 3);
    }
    v16bf a = load_a16x32(A, K, tm, k);            // reused by 4 WMMAs
    a0 = wmma_bf16(a, load_bn(BT, K, k, tn),      a0);
    a1 = wmma_bf16(a, load_bn(BT, K, k, tn + 16), a1);
    a2 = wmma_bf16(a, load_bn(BT, K, k, tn + 32), a2);
    a3 = wmma_bf16(a, load_bn(BT, K, k, tn + 48), a3);
  }
  int lane = threadIdx.x & 31;
  int nn = lane & 15;
  int mb = tm + ((lane >> 4) << 3);
  v8f accs[4] = {a0, a1, a2, a3};
#pragma unroll
  for (int j = 0; j < 4; ++j) {
#pragma unroll
    for (int r = 0; r < 8; ++r) {
      int n = tn + j * 16 + nn;
      if constexpr (TC) C[(size_t)n * M + mb + r]  = f2bf(accs[j][r]);   // C^T (N x M)
      else              C[(size_t)(mb + r) * Nn + n] = f2bf(accs[j][r]); // C (M x N)
    }
  }
}

// ---------------- block-sparse flash attention ------------------------------
// grid = Nb blocks, 256 threads = 8 waves, wave w = head w of query block b.
// km is (KV x D) row-major (contiguous along dh); vmT is (D x KV) (contiguous along key).
__global__ void attention(const bf16* __restrict__ qm, const bf16* __restrict__ km,
                          const bf16* __restrict__ vmT, const int* __restrict__ anchors,
                          const unsigned char* __restrict__ keep, bf16* __restrict__ attn) {
  __shared__ bf16 pLds[8][16 * 32];
  const int qb   = blockIdx.x;
  const int h    = threadIdx.x >> 5;
  const int lane = threadIdx.x & 31;
  const int q0   = qb << 4;
  const int nctx = keep[qb] ? anchors[qb] : 0;   // mask limit, uniform per block
  const float SCALE = 0.125f;                    // 1/sqrt(64)

  v16bf qa0 = load_a16x32(qm, D, q0, h * DH);
  v16bf qa1 = load_a16x32(qm, D, q0, h * DH + 32);

  float m_i[8], l_i[8];
  v8f o0 = zero8(), o1 = zero8(), o2 = zero8(), o3 = zero8();
#pragma unroll
  for (int r = 0; r < 8; ++r) { m_i[r] = -1e30f; l_i[r] = 0.f; }

  auto process = [&](auto clampc, int key0, int vlimit) {
    constexpr bool CL = decltype(clampc)::value;
    // scores for 2 x 16 keys: K^T fragments = load_bn with n = key index
    v8f s0 = zero8(), s1 = zero8();
    if constexpr (CL) {
      s0 = wmma_bf16(qa0, load_bn_clampN(km, D, h * DH,      key0,      KV), s0);
      s0 = wmma_bf16(qa1, load_bn_clampN(km, D, h * DH + 32, key0,      KV), s0);
      s1 = wmma_bf16(qa0, load_bn_clampN(km, D, h * DH,      key0 + 16, KV), s1);
      s1 = wmma_bf16(qa1, load_bn_clampN(km, D, h * DH + 32, key0 + 16, KV), s1);
    } else {
      s0 = wmma_bf16(qa0, load_bn(km, D, h * DH,      key0),      s0);
      s0 = wmma_bf16(qa1, load_bn(km, D, h * DH + 32, key0),      s0);
      s1 = wmma_bf16(qa0, load_bn(km, D, h * DH,      key0 + 16), s1);
      s1 = wmma_bf16(qa1, load_bn(km, D, h * DH + 32, key0 + 16), s1);
    }
    const bool v0 = (key0 +      (lane & 15)) < vlimit;
    const bool v1 = (key0 + 16 + (lane & 15)) < vlimit;
    float p0[8], p1[8], sc[8];
#pragma unroll
    for (int r = 0; r < 8; ++r) {
      float a = v0 ? s0[r] * SCALE : -1e9f;
      float b = v1 ? s1[r] * SCALE : -1e9f;
      float cm = fmaxf(a, b);
#pragma unroll
      for (int off = 8; off; off >>= 1) cm = fmaxf(cm, __shfl_xor(cm, off, 32));
      float mn = fmaxf(m_i[r], cm);
      sc[r] = __expf(m_i[r] - mn);
      p0[r] = __expf(a - mn);
      p1[r] = __expf(b - mn);
      float rs = p0[r] + p1[r];
#pragma unroll
      for (int off = 8; off; off >>= 1) rs += __shfl_xor(rs, off, 32);
      l_i[r] = l_i[r] * sc[r] + rs;
      m_i[r] = mn;
    }
#pragma unroll
    for (int r = 0; r < 8; ++r) { o0[r] *= sc[r]; o1[r] *= sc[r]; o2[r] *= sc[r]; o3[r] *= sc[r]; }
    // P (C-layout) -> LDS -> A-layout fragment
    bf16* pl = pLds[h];
    int col = lane & 15, mb = (lane >> 4) << 3;
#pragma unroll
    for (int r = 0; r < 8; ++r) {
      pl[(mb + r) * 32 + col]      = f2bf(p0[r]);
      pl[(mb + r) * 32 + col + 16] = f2bf(p1[r]);
    }
    __syncthreads();
    v16bf pa = load_a16x32(pl, 32, 0, 0);
    __syncthreads();
    // V^T fragments: contiguous along key
    if constexpr (CL) {
      o0 = wmma_bf16(pa, load_bn_clampK(vmT, KV, key0, h * DH,      KV), o0);
      o1 = wmma_bf16(pa, load_bn_clampK(vmT, KV, key0, h * DH + 16, KV), o1);
      o2 = wmma_bf16(pa, load_bn_clampK(vmT, KV, key0, h * DH + 32, KV), o2);
      o3 = wmma_bf16(pa, load_bn_clampK(vmT, KV, key0, h * DH + 48, KV), o3);
    } else {
      o0 = wmma_bf16(pa, load_bn(vmT, KV, key0, h * DH),      o0);
      o1 = wmma_bf16(pa, load_bn(vmT, KV, key0, h * DH + 16), o1);
      o2 = wmma_bf16(pa, load_bn(vmT, KV, key0, h * DH + 32), o2);
      o3 = wmma_bf16(pa, load_bn(vmT, KV, key0, h * DH + 48), o3);
    }
  };

  for (int c = 0; c < nctx; c += 32) process(BoolC<false>{}, c, nctx);   // context
  process(BoolC<true>{}, S + qb * 16, S + qb * 16 + 16);                 // own draft block

  int col = lane & 15, mb = (lane >> 4) << 3;
#pragma unroll
  for (int r = 0; r < 8; ++r) {
    float inv = 1.f / fmaxf(l_i[r], 1e-20f);
    size_t row = (size_t)(q0 + mb + r) * D + h * DH + col;
    attn[row]      = f2bf(o0[r] * inv);
    attn[row + 16] = f2bf(o1[r] * inv);
    attn[row + 32] = f2bf(o2[r] * inv);
    attn[row + 48] = f2bf(o3[r] * inv);
  }
}

// ------- LM head: 16x256 logit tiles (4 waves x 16x64) + softmax partials ---
__global__ void logits_partial(const bf16* __restrict__ outo, const bf16* __restrict__ wlmT,
                               float4* __restrict__ partials) {
  __shared__ float lg[16 * VB];
  int vb = blockIdx.x;                 // 0..124
  int rt = blockIdx.y;                 // 0..127
  int w = threadIdx.x >> 5, lane = threadIdx.x & 31;
  int m0 = rt << 4;
  int n0 = vb * VB + w * 64;
  v8f a0 = zero8(), a1 = zero8(), a2 = zero8(), a3 = zero8();
  for (int k = 0; k < D; k += 32) {
    if (k + 32 < D) {
      __builtin_prefetch(wlmT + (size_t)(n0 + (threadIdx.x & 31)) * D + k + 32, 0, 3);
      __builtin_prefetch(wlmT + (size_t)(n0 + 32 + (threadIdx.x & 31)) * D + k + 32, 0, 3);
    }
    v16bf a = load_a16x32(outo, D, m0, k);
    a0 = wmma_bf16(a, load_bn(wlmT, D, k, n0),      a0);
    a1 = wmma_bf16(a, load_bn(wlmT, D, k, n0 + 16), a1);
    a2 = wmma_bf16(a, load_bn(wlmT, D, k, n0 + 32), a2);
    a3 = wmma_bf16(a, load_bn(wlmT, D, k, n0 + 48), a3);
  }
  int nn = lane & 15, mb = (lane >> 4) << 3;
  v8f accs[4] = {a0, a1, a2, a3};
#pragma unroll
  for (int j = 0; j < 4; ++j)
#pragma unroll
    for (int r = 0; r < 8; ++r)
      lg[(mb + r) * VB + w * 64 + j * 16 + nn] = accs[j][r];
  __syncthreads();
  if (threadIdx.x < 16) {
    const float* rowp = lg + threadIdx.x * VB;
    float mx = rowp[0]; int bi = 0;
    for (int c = 1; c < VB; ++c) { float v = rowp[c]; if (v > mx) { mx = v; bi = c; } }
    float sum = 0.f;
    for (int c = 0; c < VB; ++c) sum += __expf(rowp[c] - mx);
    float4 p; p.x = mx; p.y = sum; p.z = __int_as_float(vb * VB + bi); p.w = 0.f;
    partials[(size_t)(m0 + threadIdx.x) * NVB + vb] = p;
  }
}

__global__ void target_logit_k(const bf16* __restrict__ outo, const bf16* __restrict__ wlmT,
                               const int* __restrict__ target, float* __restrict__ tlogit) {
  int row = (blockIdx.x * blockDim.x + threadIdx.x) >> 5;
  int lane = threadIdx.x & 31;
  if (row >= Q) return;
  int t = target[row];
  float s = 0.f;
  for (int k = lane; k < D; k += 32)
    s += bf2f(outo[(size_t)row * D + k]) * bf2f(wlmT[(size_t)t * D + k]);  // both contiguous
#pragma unroll
  for (int off = 16; off; off >>= 1) s += __shfl_xor(s, off, 32);
  if (lane == 0) tlogit[row] = s;
}

__global__ void zero_acc(float* a) { if (threadIdx.x < 4) a[threadIdx.x] = 0.f; }

__global__ void row_reduce(const float4* __restrict__ partials, const float* __restrict__ tlogit,
                           const int* __restrict__ target, const float* __restrict__ weight,
                           float* __restrict__ acc) {
  int r = blockIdx.x * blockDim.x + threadIdx.x;
  if (r >= Q) return;
  const float4* p = partials + (size_t)r * NVB;
  float M = -1e30f;
  for (int i = 0; i < NVB; ++i) M = fmaxf(M, p[i].x);
  float sum = 0.f, bv = -1e30f; int bi = 0;
  for (int i = 0; i < NVB; ++i) {
    float4 t = p[i];
    sum += t.y * __expf(t.x - M);
    if (t.x > bv) { bv = t.x; bi = __float_as_int(t.z); }
  }
  float lse = M + __logf(sum);
  float lp  = lse - tlogit[r];
  float w   = weight[r];
  atomicAdd(&acc[0], w * lp);
  atomicAdd(&acc[1], w);
  atomicAdd(&acc[2], ((bi == target[r]) && (w > 0.5f)) ? 1.f : 0.f);
}

__global__ void finalize(const float* __restrict__ a, float* __restrict__ out) {
  if (threadIdx.x == 0) {
    out[0] = a[0] / (a[1] + 1e-6f);
    out[1] = a[2] / (a[1] + 1e-6f);
  }
}

// ---------------- launch ----------------------------------------------------
extern "C" void kernel_launch(void* const* d_in, const int* in_sizes, int n_in,
                              void* d_out, int out_size, void* d_ws, size_t ws_size,
                              hipStream_t stream) {
  const int*   ids     = (const int*)d_in[0];
  const float* hidden  = (const float*)d_in[1];
  const float* lmask   = (const float*)d_in[2];
  const int*   anchors = (const int*)d_in[3];
  const unsigned char* keep = (const unsigned char*)d_in[4];
  const float* embed   = (const float*)d_in[5];
  const float* Wq = (const float*)d_in[6];
  const float* Wk = (const float*)d_in[7];
  const float* Wv = (const float*)d_in[8];
  const float* Wo = (const float*)d_in[9];
  const float* Wlm = (const float*)d_in[10];
  float* out = (float*)d_out;

  char* p = (char*)d_ws;
  auto take = [&](size_t bytes) { char* r = p; p += (bytes + 255) & ~(size_t)255; return r; };
  bf16* kv    = (bf16*)take((size_t)KV * D * 2);     // [hidden_bf16 ; noise_embed_bf16]
  bf16* wqT   = (bf16*)take((size_t)D * D * 2);      // transposed weights (N-major)
  bf16* wkT   = (bf16*)take((size_t)D * D * 2);
  bf16* wvT   = (bf16*)take((size_t)D * D * 2);
  bf16* woT   = (bf16*)take((size_t)D * D * 2);
  bf16* wlmT  = (bf16*)take((size_t)D * V * 2);      // (V x D)
  bf16* qm    = (bf16*)take((size_t)Q * D * 2);
  bf16* km    = (bf16*)take((size_t)KV * D * 2);     // (KV x D)
  bf16* vmT   = (bf16*)take((size_t)KV * D * 2);     // (D x KV)
  bf16* attnb = (bf16*)take((size_t)Q * D * 2);
  bf16* outo  = (bf16*)take((size_t)Q * D * 2);
  float4* parts = (float4*)take((size_t)Q * NVB * 16);
  int*   target = (int*)take((size_t)Q * 4);
  float* weight = (float*)take((size_t)Q * 4);
  float* tlog   = (float*)take((size_t)Q * 4);
  float* acc    = (float*)take(4 * 4);

  const int T = 256;
  // prep: bf16 conversion (+ transposition of all B operands)
  cvt_f32_bf16<<<(S * D + T - 1) / T, T, 0, stream>>>(hidden, kv, S * D);
  embed_noise<<<(Q * D + T - 1) / T, T, 0, stream>>>(ids, anchors, keep, embed, kv + (size_t)S * D);
  cvt_transpose<<<dim3(D / 32, D / 32), T, 0, stream>>>(Wq, wqT, D, D);
  cvt_transpose<<<dim3(D / 32, D / 32), T, 0, stream>>>(Wk, wkT, D, D);
  cvt_transpose<<<dim3(D / 32, D / 32), T, 0, stream>>>(Wv, wvT, D, D);
  cvt_transpose<<<dim3(D / 32, D / 32), T, 0, stream>>>(Wo, woT, D, D);
  cvt_transpose<<<dim3(D / 32, V / 32), T, 0, stream>>>(Wlm, wlmT, D, V);
  targets_weights<<<(Q + T - 1) / T, T, 0, stream>>>(ids, lmask, anchors, keep, target, weight);

  // projections (WMMA, 16x64 wave tiles); V written transposed for attention
  gemm_bf16<false><<<(Q  / 16) * (D / 64) / 8, T, 0, stream>>>(kv + (size_t)S * D, wqT, qm, Q, D, D);
  gemm_bf16<false><<<(KV / 16) * (D / 64) / 8, T, 0, stream>>>(kv, wkT, km, KV, D, D);
  gemm_bf16<true ><<<(KV / 16) * (D / 64) / 8, T, 0, stream>>>(kv, wvT, vmT, KV, D, D);

  // block-sparse flash attention (WMMA)
  attention<<<Nb, T, 0, stream>>>(qm, km, vmT, anchors, keep, attnb);

  // output projection (WMMA)
  gemm_bf16<false><<<(Q / 16) * (D / 64) / 8, T, 0, stream>>>(attnb, woT, outo, Q, D, D);

  // LM head + fused softmax stats (WMMA), then reductions
  logits_partial<<<dim3(NVB, Q / 16), 128, 0, stream>>>(outo, wlmT, parts);
  target_logit_k<<<(Q * 32) / T, T, 0, stream>>>(outo, wlmT, target, tlog);
  zero_acc<<<1, 32, 0, stream>>>(acc);
  row_reduce<<<Q / T, T, 0, stream>>>(parts, tlog, target, weight, acc);
  finalize<<<1, 32, 0, stream>>>(acc, out);
  (void)in_sizes; (void)n_in; (void)out_size; (void)ws_size;
}
